// GNNPredictor_5506148073958
// MI455X (gfx1250) — matrix operand order repaired and checked by reading, not compiled, not run on hardware
//
#include <hip/hip_runtime.h>
#include <hip/hip_bf16.h>

// ---------------------------------------------------------------------------
// CDNA5 (gfx1250) bipartite-GNN forward. wave32, v_wmma_f32_16x16x32_bf16.
// Weights pre-swizzled into bf16 B-fragment layout (32B/lane vector loads).
// ---------------------------------------------------------------------------

typedef __attribute__((ext_vector_type(16))) __bf16        v16bf;
typedef __attribute__((ext_vector_type(8)))  float         v8f;
typedef __attribute__((ext_vector_type(8)))  unsigned int  v8u;

#define NNODE 50000
#define NE    600000
#define EMB   64

__device__ __forceinline__ v8f wmma_bf16(v16bf a, v16bf b, v8f c) {
  return __builtin_amdgcn_wmma_f32_16x16x32_bf16(false, a, false, b, (short)0, c,
                                                 false, false);
}

// A fragment (16x32 bf16) from an f32 tile in LDS, row stride `stride`.
// Layout (contiguous k order used here, same as the B swizzle): lane m=lane&15, hi=lane>>4; vgpr j -> K = 16*hi+2*j.
__device__ __forceinline__ v16bf a_frag(const float* base, int stride, int kb, int lane) {
  int m = lane & 15, hi = lane >> 4;
  const float* rp = base + m * stride + kb;
  v16bf a;
#pragma unroll
  for (int j = 0; j < 8; ++j) {
    int k = 16 * hi + 2 * j;
    a[2 * j]     = (__bf16)rp[k];
    a[2 * j + 1] = (__bf16)rp[k + 1];
  }
  return a;
}

// B fragment from pre-swizzled weights: one 32-byte contiguous load per lane.
__device__ __forceinline__ v16bf b_frag_sw(const unsigned int* __restrict__ buf,
                                           int frag, int lane) {
  v8u r = *(const v8u*)(buf + ((size_t)frag * 32 + lane) * 8);
  return __builtin_bit_cast(v16bf, r);
}

// ---------------------------------------------------------------------------
// Weight swizzle pre-pass: f32 W[K][ld] (valid rows < Kmax) -> bf16 fragments.
// frag f = kt*(N/16)+nt ; dword d in frag: lane=(d>>3)&31, j=d&7;
// element pair K = kt*32 + 16*(lane>>4) + 2*j, N = nt*16 + (lane&15).
// ---------------------------------------------------------------------------
__global__ void swizzle_b_kernel(const float* __restrict__ W, int ld, int Kpad, int N,
                                 int Kmax, unsigned int* __restrict__ out) {
  int ntiles = N >> 4;
  int total = (Kpad >> 5) * ntiles * 256;
  for (int d = blockIdx.x * 256 + threadIdx.x; d < total; d += gridDim.x * 256) {
    int j = d & 7;
    int lane = (d >> 3) & 31;
    int f = d >> 8;
    int kt = f / ntiles, nt = f % ntiles;
    int hi = lane >> 4, n = nt * 16 + (lane & 15);
    int k = kt * 32 + 16 * hi + 2 * j;
    float x0 = (k     < Kmax) ? W[(size_t)k * ld + n]       : 0.f;
    float x1 = (k + 1 < Kmax) ? W[(size_t)(k + 1) * ld + n] : 0.f;
    __bf16 b0 = (__bf16)x0, b1 = (__bf16)x1;
    unsigned int u0 = __builtin_bit_cast(unsigned short, b0);
    unsigned int u1 = __builtin_bit_cast(unsigned short, b1);
    out[d] = u0 | (u1 << 16);
  }
}

// ---------------------------------------------------------------------------
// Deterministic column reductions (sum, sumsq)
// ---------------------------------------------------------------------------
__global__ void colsum64_kernel(const float* __restrict__ x, int N, float* __restrict__ part) {
  __shared__ float ls[256], lq[256];
  int c = threadIdx.x & 63, sub = threadIdx.x >> 6;
  float s = 0.f, q = 0.f;
  for (int r = blockIdx.x * 4 + sub; r < N; r += gridDim.x * 4) {
    float v = x[(size_t)r * 64 + c];
    s += v; q += v * v;
  }
  ls[threadIdx.x] = s; lq[threadIdx.x] = q;
  __syncthreads();
  if (threadIdx.x < 64) {
    s = ls[c] + ls[c + 64] + ls[c + 128] + ls[c + 192];
    q = lq[c] + lq[c + 64] + lq[c + 128] + lq[c + 192];
    part[blockIdx.x * 128 + c]      = s;
    part[blockIdx.x * 128 + 64 + c] = q;
  }
}

// mode 0 = BatchNorm (var = E[x^2]-m^2); mode 1 = GraphNorm (E[x^2]-(2a-a^2)m^2).
__global__ void finalize_norm_kernel(const float* __restrict__ part, int B, int N, int mode,
                                     const float* __restrict__ alpha,
                                     float* __restrict__ meanOut, float* __restrict__ rsigOut) {
  int c = threadIdx.x;  // 64 threads
  float s = 0.f, q = 0.f;
  for (int b = 0; b < B; ++b) { s += part[b * 128 + c]; q += part[b * 128 + 64 + c]; }
  float m = s / (float)N, ex2 = q / (float)N;
  float var;
  if (mode == 0) var = ex2 - m * m;
  else { float al = alpha[c]; var = ex2 - (2.f * al - al * al) * m * m; }
  meanOut[c] = m;
  rsigOut[c] = rsqrtf(var + 1e-5f);
}

__global__ void colsum_small_kernel(const float* __restrict__ x, int N, int C,
                                    float* __restrict__ part) {
  float s[8], q[8];
#pragma unroll
  for (int c = 0; c < 8; ++c) { s[c] = 0.f; q[c] = 0.f; }
  for (int r = blockIdx.x * 256 + threadIdx.x; r < N; r += gridDim.x * 256)
    for (int c = 0; c < C; ++c) { float v = x[(size_t)r * C + c]; s[c] += v; q[c] += v * v; }
  __shared__ float red[256];
  for (int c = 0; c < C; ++c) {
    red[threadIdx.x] = s[c]; __syncthreads();
    for (int o = 128; o > 0; o >>= 1) {
      if (threadIdx.x < o) red[threadIdx.x] += red[threadIdx.x + o];
      __syncthreads();
    }
    if (threadIdx.x == 0) part[blockIdx.x * 2 * C + c] = red[0];
    __syncthreads();
    red[threadIdx.x] = q[c]; __syncthreads();
    for (int o = 128; o > 0; o >>= 1) {
      if (threadIdx.x < o) red[threadIdx.x] += red[threadIdx.x + o];
      __syncthreads();
    }
    if (threadIdx.x == 0) part[blockIdx.x * 2 * C + C + c] = red[0];
    __syncthreads();
  }
}

__global__ void finalize_small_kernel(const float* __restrict__ part, int B, int N, int C,
                                      float* __restrict__ meanOut, float* __restrict__ rsigOut) {
  int c = threadIdx.x;
  if (c < C) {
    float s = 0.f, q = 0.f;
    for (int b = 0; b < B; ++b) { s += part[b * 2 * C + c]; q += part[b * 2 * C + C + c]; }
    float m = s / (float)N, var = q / (float)N - m * m;
    meanOut[c] = m;
    rsigOut[c] = rsqrtf(var + 1e-5f);
  }
}

// ---------------------------------------------------------------------------
// Utility kernels
// ---------------------------------------------------------------------------
__global__ void zero_f_kernel(float* __restrict__ p, size_t n) {
  size_t i = (size_t)blockIdx.x * 256 + threadIdx.x;
  if (i < n) p[i] = 0.f;
}
__global__ void zero_i_kernel(int* __restrict__ p, int n) {
  int i = blockIdx.x * 256 + threadIdx.x;
  if (i < n) p[i] = 0;
}
__global__ void count_edges_kernel(const int* __restrict__ ci, const int* __restrict__ vi,
                                   int E, int* __restrict__ cc, int* __restrict__ cv) {
  int e = blockIdx.x * 256 + threadIdx.x;
  if (e < E) { atomicAdd(&cc[ci[e]], 1); atomicAdd(&cv[vi[e]], 1); }
}
__global__ void divide_cnt_kernel(float* __restrict__ agg, const int* __restrict__ cnt, int N) {
  int i = blockIdx.x * 256 + threadIdx.x;
  if (i < N * 64) {
    int cn = cnt[i >> 6];
    agg[i] /= (float)(cn > 0 ? cn : 1);
  }
}
__global__ void apply_gn_kernel(float* __restrict__ x, int N,
                                const float* __restrict__ meanv, const float* __restrict__ rsig,
                                const float* __restrict__ w, const float* __restrict__ b,
                                const float* __restrict__ alpha) {
  int i = blockIdx.x * 256 + threadIdx.x;
  if (i < N * 64) {
    int c = i & 63;
    x[i] = w[c] * (x[i] - alpha[c] * meanv[c]) * rsig[c] + b[c];
  }
}

// ---------------------------------------------------------------------------
// Embedding MLP: bn(x[N,C]) @ W1[C,64] -> relu -> @ W2[64,64] -> relu
// 16 rows/block, 128 threads (4 waves x one 16-col tile).
// ---------------------------------------------------------------------------
__global__ void embed_kernel(const float* __restrict__ x, int C,
                             const float* __restrict__ nm, const float* __restrict__ nr,
                             const float* __restrict__ bw, const float* __restrict__ bb,
                             const unsigned int* __restrict__ W1sw, const float* __restrict__ b1,
                             const unsigned int* __restrict__ W2sw, const float* __restrict__ b2,
                             float* __restrict__ out) {
  __shared__ float sX[16 * 36];
  __shared__ float sH[16 * 68];
  int r0 = blockIdx.x * 16;
  for (int i = threadIdx.x; i < 16 * 32; i += blockDim.x) {
    int row = i >> 5, col = i & 31;
    float v = 0.f;
    if (col < C) {
      float raw = x[(size_t)(r0 + row) * C + col];
      v = (raw - nm[col]) * nr[col] * bw[col] + bb[col];
    }
    sX[row * 36 + col] = v;
  }
  __syncthreads();
  int wid = threadIdx.x >> 5, lane = threadIdx.x & 31;
  int n = wid * 16 + (lane & 15), hi = lane >> 4;
  v8f acc = {};
  acc = wmma_bf16(a_frag(sX, 36, 0, lane), b_frag_sw(W1sw, wid, lane), acc);
  float bias1 = b1[n];
#pragma unroll
  for (int r = 0; r < 8; ++r) sH[(r + 8 * hi) * 68 + n] = fmaxf(acc[r] + bias1, 0.f);
  __syncthreads();
  v8f acc2 = {};
  acc2 = wmma_bf16(a_frag(sH, 68, 0, lane),  b_frag_sw(W2sw, 0 * 4 + wid, lane), acc2);
  acc2 = wmma_bf16(a_frag(sH, 68, 32, lane), b_frag_sw(W2sw, 1 * 4 + wid, lane), acc2);
  float bias2 = b2[n];
#pragma unroll
  for (int r = 0; r < 8; ++r)
    out[(size_t)(r0 + r + 8 * hi) * 64 + n] = fmaxf(acc2[r] + bias2, 0.f);
}

// ---------------------------------------------------------------------------
// Fused edge message MLP + scatter-add.
// msg = relu([right[dst] | e | left[src]] @ W1 + b1) @ W2 + b2 ; agg[dst] += msg
// K=129 split: 64 (right) + 64 (left) via WMMA, edge scalar as rank-1 VALU term.
// ---------------------------------------------------------------------------
__global__ void edge_conv_kernel(const float* __restrict__ left, const float* __restrict__ right,
                                 const float* __restrict__ eraw, const float* __restrict__ estat,
                                 const float* __restrict__ ebw, const float* __restrict__ ebb,
                                 const int* __restrict__ sidx, const int* __restrict__ didx,
                                 const unsigned int* __restrict__ W1Rsw,
                                 const unsigned int* __restrict__ W1Lsw,
                                 const float* __restrict__ Wedge,  // fmW1 row 64 (f32[64])
                                 const float* __restrict__ b1,
                                 const unsigned int* __restrict__ W2sw,
                                 const float* __restrict__ b2,
                                 float* __restrict__ agg) {
  __shared__ int   sD[16], sS[16];
  __shared__ float sE[16];
  __shared__ float sR[16 * 68], sL[16 * 68], sH[16 * 68];
  int e0 = blockIdx.x * 16;
  int t = threadIdx.x;
  if (t < 16) {
    sD[t] = didx[e0 + t];
    sS[t] = sidx[e0 + t];
    sE[t] = (eraw[e0 + t] - estat[0]) * estat[1] * ebw[0] + ebb[0];
  }
  __syncthreads();
  // Stage gathered endpoint rows into LDS (16B vector loads, rows are L2-resident).
  for (int i = t; i < 16 * 16; i += blockDim.x) {
    int row = i >> 4, c4 = (i & 15) << 2;
    float4 vR = *(const float4*)(right + (size_t)sD[row] * 64 + c4);
    float4 vL = *(const float4*)(left  + (size_t)sS[row] * 64 + c4);
    *(float4*)(sR + row * 68 + c4) = vR;
    *(float4*)(sL + row * 68 + c4) = vL;
  }
  __syncthreads();
  int wid = t >> 5, lane = t & 31;
  int n = wid * 16 + (lane & 15), hi = lane >> 4;
  v8f acc = {};
  acc = wmma_bf16(a_frag(sR, 68, 0, lane),  b_frag_sw(W1Rsw, 0 * 4 + wid, lane), acc);
  acc = wmma_bf16(a_frag(sR, 68, 32, lane), b_frag_sw(W1Rsw, 1 * 4 + wid, lane), acc);
  acc = wmma_bf16(a_frag(sL, 68, 0, lane),  b_frag_sw(W1Lsw, 0 * 4 + wid, lane), acc);
  acc = wmma_bf16(a_frag(sL, 68, 32, lane), b_frag_sw(W1Lsw, 1 * 4 + wid, lane), acc);
  float we = Wedge[n];
  float bias1 = b1[n];
#pragma unroll
  for (int r = 0; r < 8; ++r) {
    int m = r + 8 * hi;
    sH[m * 68 + n] = fmaxf(acc[r] + bias1 + sE[m] * we, 0.f);
  }
  __syncthreads();
  v8f acc2 = {};
  acc2 = wmma_bf16(a_frag(sH, 68, 0, lane),  b_frag_sw(W2sw, 0 * 4 + wid, lane), acc2);
  acc2 = wmma_bf16(a_frag(sH, 68, 32, lane), b_frag_sw(W2sw, 1 * 4 + wid, lane), acc2);
  float bias2 = b2[n];
#pragma unroll
  for (int r = 0; r < 8; ++r) {
    int m = r + 8 * hi;
    atomicAdd(agg + (size_t)sD[m] * 64 + n, acc2[r] + bias2);
  }
}

// ---------------------------------------------------------------------------
// Node update: relu([bn(agg) | rightOld] @ oW1 + ob1) @ oW2 + ob2 -> new right
// ---------------------------------------------------------------------------
__global__ void node_update_kernel(const float* __restrict__ agg,
                                   const float* __restrict__ meanv, const float* __restrict__ rsig,
                                   const float* __restrict__ bnw, const float* __restrict__ bnb,
                                   const float* __restrict__ rightOld,
                                   const unsigned int* __restrict__ W1sw,
                                   const float* __restrict__ b1,
                                   const unsigned int* __restrict__ W2sw,
                                   const float* __restrict__ b2,
                                   float* __restrict__ outp) {
  __shared__ float sX[16 * 132], sH[16 * 68];
  int r0 = blockIdx.x * 16;
  for (int i = threadIdx.x; i < 16 * 128; i += blockDim.x) {
    int row = i >> 7, col = i & 127;
    float v;
    if (col < 64)
      v = (agg[(size_t)(r0 + row) * 64 + col] - meanv[col]) * rsig[col] * bnw[col] + bnb[col];
    else
      v = rightOld[(size_t)(r0 + row) * 64 + (col - 64)];
    sX[row * 132 + col] = v;
  }
  __syncthreads();
  int wid = threadIdx.x >> 5, lane = threadIdx.x & 31;
  int n = wid * 16 + (lane & 15), hi = lane >> 4;
  v8f acc = {};
#pragma unroll
  for (int kt = 0; kt < 4; ++kt)
    acc = wmma_bf16(a_frag(sX, 132, kt * 32, lane), b_frag_sw(W1sw, kt * 4 + wid, lane), acc);
  float bias1 = b1[n];
#pragma unroll
  for (int r = 0; r < 8; ++r) sH[(r + 8 * hi) * 68 + n] = fmaxf(acc[r] + bias1, 0.f);
  __syncthreads();
  v8f acc2 = {};
  acc2 = wmma_bf16(a_frag(sH, 68, 0, lane),  b_frag_sw(W2sw, 0 * 4 + wid, lane), acc2);
  acc2 = wmma_bf16(a_frag(sH, 68, 32, lane), b_frag_sw(W2sw, 1 * 4 + wid, lane), acc2);
  float bias2 = b2[n];
#pragma unroll
  for (int r = 0; r < 8; ++r)
    outp[(size_t)(r0 + r + 8 * hi) * 64 + n] = acc2[r] + bias2;  // no relu here
}

// ---------------------------------------------------------------------------
// Final MLP: [v0|v1|v2] (192) -> 512 relu -> 128 relu -> 1
// 16 rows/block, 256 threads (8 waves). h1/h2 staged in LDS.
// ---------------------------------------------------------------------------
__global__ void final_mlp_kernel(const float* __restrict__ v0, const float* __restrict__ v1,
                                 const float* __restrict__ v2,
                                 const unsigned int* __restrict__ W1sw,
                                 const float* __restrict__ b1,
                                 const unsigned int* __restrict__ W2sw,
                                 const float* __restrict__ b2,
                                 const float* __restrict__ W3, const float* __restrict__ b3,
                                 float* __restrict__ outp) {
  __shared__ float sIn[16 * 196];
  __shared__ float sH1[16 * 516];
  __shared__ float sH2[16 * 132];
  int r0 = blockIdx.x * 16;
  for (int i = threadIdx.x; i < 16 * 192; i += blockDim.x) {
    int row = i / 192, col = i % 192;
    const float* src = (col < 64) ? v0 : (col < 128 ? v1 : v2);
    sIn[row * 196 + col] = src[(size_t)(r0 + row) * 64 + (col & 63)];
  }
  __syncthreads();
  int wid = threadIdx.x >> 5, lane = threadIdx.x & 31;
  int hi = lane >> 4, nl = lane & 15;
  // Layer 1: N=512 (8 waves x 4 N-tiles), K=192 (6 K-tiles), ntiles=32
  v8f acc[4] = {};
#pragma unroll
  for (int kt = 0; kt < 6; ++kt) {
    v16bf a = a_frag(sIn, 196, kt * 32, lane);
#pragma unroll
    for (int i2 = 0; i2 < 4; ++i2)
      acc[i2] = wmma_bf16(a, b_frag_sw(W1sw, kt * 32 + wid * 4 + i2, lane), acc[i2]);
  }
#pragma unroll
  for (int i2 = 0; i2 < 4; ++i2) {
    int n = (wid * 4 + i2) * 16 + nl;
    float bias = b1[n];
#pragma unroll
    for (int r = 0; r < 8; ++r)
      sH1[(r + 8 * hi) * 516 + n] = fmaxf(acc[i2][r] + bias, 0.f);
  }
  __syncthreads();
  // Layer 2: N=128 (8 waves x 1 tile), K=512 (16 K-tiles), ntiles=8
  v8f acc2 = {};
#pragma unroll
  for (int kt = 0; kt < 16; ++kt)
    acc2 = wmma_bf16(a_frag(sH1, 516, kt * 32, lane), b_frag_sw(W2sw, kt * 8 + wid, lane), acc2);
  int n2 = wid * 16 + nl;
  float bias2 = b2[n2];
#pragma unroll
  for (int r = 0; r < 8; ++r)
    sH2[(r + 8 * hi) * 132 + n2] = fmaxf(acc2[r] + bias2, 0.f);
  __syncthreads();
  // Layer 3: 128-dot per row
  if (threadIdx.x < 16) {
    float s = b3[0];
    for (int k = 0; k < 128; ++k) s += sH2[threadIdx.x * 132 + k] * W3[k];
    outp[r0 + threadIdx.x] = s;
  }
}

// ---------------------------------------------------------------------------
// Host orchestration
// ---------------------------------------------------------------------------
extern "C" void kernel_launch(void* const* d_in, const int* in_sizes, int n_in,
                              void* d_out, int out_size, void* d_ws, size_t ws_size,
                              hipStream_t stream) {
  const float* consF = (const float*)d_in[0];
  const float* eattr = (const float*)d_in[1];
  const float* varF  = (const float*)d_in[2];
  const float* cbnw  = (const float*)d_in[3];
  const float* cbnb  = (const float*)d_in[4];
  const float* cW1   = (const float*)d_in[5];
  const float* cb1   = (const float*)d_in[6];
  const float* cW2   = (const float*)d_in[7];
  const float* cb2   = (const float*)d_in[8];
  const float* ebnw  = (const float*)d_in[9];
  const float* ebnb  = (const float*)d_in[10];
  const float* vbnw  = (const float*)d_in[11];
  const float* vbnb  = (const float*)d_in[12];
  const float* vW1   = (const float*)d_in[13];
  const float* vb1   = (const float*)d_in[14];
  const float* vW2   = (const float*)d_in[15];
  const float* vb2   = (const float*)d_in[16];
  const float* fmW1  = (const float*)d_in[17];  // [4][129][64]
  const float* fmb1  = (const float*)d_in[18];  // [4][64]
  const float* fmW2  = (const float*)d_in[19];  // [4][64][64]
  const float* fmb2  = (const float*)d_in[20];
  const float* bnw   = (const float*)d_in[21];
  const float* bnb   = (const float*)d_in[22];
  const float* omW1  = (const float*)d_in[23];  // [4][128][64]
  const float* omb1  = (const float*)d_in[24];
  const float* omW2  = (const float*)d_in[25];  // [4][64][64]
  const float* omb2  = (const float*)d_in[26];
  const float* gnW   = (const float*)d_in[27];
  const float* gnB   = (const float*)d_in[28];
  const float* gnA   = (const float*)d_in[29];
  const float* oW1   = (const float*)d_in[30];  // [192][512]
  const float* ob1   = (const float*)d_in[31];
  const float* oW2   = (const float*)d_in[32];  // [512][128]
  const float* ob2   = (const float*)d_in[33];
  const float* oW3   = (const float*)d_in[34];  // [128][1]
  const float* ob3   = (const float*)d_in[35];
  const int*   eidx  = (const int*)d_in[36];    // [2][NE]
  const int*   eC    = eidx;                    // constraint indices (row 0)
  const int*   eV    = eidx + NE;               // variable indices  (row 1)

  // ---- workspace carve ----
  char* wp = (char*)d_ws;
  auto carve = [&](size_t bytes) {
    void* p = (void*)wp;
    wp += (bytes + 255) & ~(size_t)255;
    return p;
  };
  const size_t NB = (size_t)NNODE * 64 * sizeof(float);
  float* cfA  = (float*)carve(NB);
  float* cfB  = (float*)carve(NB);
  float* vfA  = (float*)carve(NB);
  float* vfB  = (float*)carve(NB);
  float* vf0  = (float*)carve(NB);
  float* agg  = (float*)carve(NB);
  int*   cntC = (int*)carve(NNODE * sizeof(int));
  int*   cntV = (int*)carve(NNODE * sizeof(int));
  float* part = (float*)carve(128 * 128 * sizeof(float));
  float* meanv = (float*)carve(64 * sizeof(float));
  float* rsigv = (float*)carve(64 * sizeof(float));
  float* consStat = (float*)carve(8 * sizeof(float));   // mean[3], rsig[3]
  float* varStat  = (float*)carve(12 * sizeof(float));  // mean[5], rsig[5]
  float* edgeStat = (float*)carve(4 * sizeof(float));   // mean[1], rsig[1]

  // Swizzled weight buffers (dword counts: (Kpad/32)*(N/16)*256)
  unsigned int* swEdgeW1R[4]; unsigned int* swEdgeW1L[4]; unsigned int* swEdgeW2[4];
  unsigned int* swOmW1[4];    unsigned int* swOmW2[4];
  for (int l = 0; l < 4; ++l) {
    swEdgeW1R[l] = (unsigned int*)carve(2048 * 4);
    swEdgeW1L[l] = (unsigned int*)carve(2048 * 4);
    swEdgeW2[l]  = (unsigned int*)carve(2048 * 4);
    swOmW1[l]    = (unsigned int*)carve(4096 * 4);
    swOmW2[l]    = (unsigned int*)carve(2048 * 4);
  }
  unsigned int* swCW1 = (unsigned int*)carve(1024 * 4);
  unsigned int* swCW2 = (unsigned int*)carve(2048 * 4);
  unsigned int* swVW1 = (unsigned int*)carve(1024 * 4);
  unsigned int* swVW2 = (unsigned int*)carve(2048 * 4);
  unsigned int* swOW1 = (unsigned int*)carve(49152 * 4);
  unsigned int* swOW2 = (unsigned int*)carve(32768 * 4);

  const int RB = 128;  // reduction stage-1 blocks
  const int nodeElems = NNODE * 64;
  const int gridElems = (nodeElems + 255) / 256;

  // ---- weight swizzle pre-pass ----
  auto swz = [&](const float* W, int ld, int Kpad, int N, int Kmax, unsigned int* out) {
    int total = (Kpad >> 5) * (N >> 4) * 256;
    swizzle_b_kernel<<<(total + 255) / 256, 256, 0, stream>>>(W, ld, Kpad, N, Kmax, out);
  };
  for (int l = 0; l < 4; ++l) {
    const float* w1 = fmW1 + (size_t)l * 129 * 64;
    swz(w1,            64, 64, 64, 64, swEdgeW1R[l]);   // rows 0..63  (right part)
    swz(w1 + 65 * 64,  64, 64, 64, 64, swEdgeW1L[l]);   // rows 65..128 (left part)
    swz(fmW2 + (size_t)l * 4096, 64, 64, 64, 64, swEdgeW2[l]);
    swz(omW1 + (size_t)l * 8192, 64, 128, 64, 128, swOmW1[l]);
    swz(omW2 + (size_t)l * 4096, 64, 64, 64, 64, swOmW2[l]);
  }
  swz(cW1, 64, 32, 64, 3, swCW1);
  swz(cW2, 64, 64, 64, 64, swCW2);
  swz(vW1, 64, 32, 64, 5, swVW1);
  swz(vW2, 64, 64, 64, 64, swVW2);
  swz(oW1, 512, 192, 512, 192, swOW1);
  swz(oW2, 128, 512, 128, 512, swOW2);

  // ---- input BatchNorm statistics (deterministic two-stage) ----
  colsum_small_kernel<<<RB, 256, 0, stream>>>(consF, NNODE, 3, part);
  finalize_small_kernel<<<1, 64, 0, stream>>>(part, RB, NNODE, 3, consStat, consStat + 3);
  colsum_small_kernel<<<RB, 256, 0, stream>>>(varF, NNODE, 5, part);
  finalize_small_kernel<<<1, 64, 0, stream>>>(part, RB, NNODE, 5, varStat, varStat + 5);
  colsum_small_kernel<<<RB, 256, 0, stream>>>(eattr, NE, 1, part);
  finalize_small_kernel<<<1, 64, 0, stream>>>(part, RB, NE, 1, edgeStat, edgeStat + 1);

  // ---- embeddings ----
  embed_kernel<<<NNODE / 16, 128, 0, stream>>>(consF, 3, consStat, consStat + 3, cbnw, cbnb,
                                               swCW1, cb1, swCW2, cb2, cfA);
  embed_kernel<<<NNODE / 16, 128, 0, stream>>>(varF, 5, varStat, varStat + 5, vbnw, vbnb,
                                               swVW1, vb1, swVW2, vb2, vfA);
  hipMemcpyAsync(vf0, vfA, NB, hipMemcpyDeviceToDevice, stream);  // save vars_outputs[0]

  // ---- degree counts (fixed over all convs) ----
  zero_i_kernel<<<(NNODE + 255) / 256, 256, 0, stream>>>(cntC, NNODE);
  zero_i_kernel<<<(NNODE + 255) / 256, 256, 0, stream>>>(cntV, NNODE);
  count_edges_kernel<<<(NE + 255) / 256, 256, 0, stream>>>(eC, eV, NE, cntC, cntV);

  // ---- conv helper ----
  auto run_conv = [&](const float* leftP, const float* rightP, const int* sIdx,
                      const int* dIdx, const int* cnt, int layer, float* outP) {
    zero_f_kernel<<<gridElems, 256, 0, stream>>>(agg, (size_t)nodeElems);
    edge_conv_kernel<<<NE / 16, 128, 0, stream>>>(
        leftP, rightP, eattr, edgeStat, ebnw, ebnb, sIdx, dIdx,
        swEdgeW1R[layer], swEdgeW1L[layer],
        fmW1 + (size_t)layer * 129 * 64 + 64 * 64,  // edge-scalar weight row
        fmb1 + layer * 64, swEdgeW2[layer], fmb2 + layer * 64, agg);
    divide_cnt_kernel<<<gridElems, 256, 0, stream>>>(agg, cnt, NNODE);
    colsum64_kernel<<<RB, 256, 0, stream>>>(agg, NNODE, part);
    finalize_norm_kernel<<<1, 64, 0, stream>>>(part, RB, NNODE, 0, gnA, meanv, rsigv);
    node_update_kernel<<<NNODE / 16, 128, 0, stream>>>(
        agg, meanv, rsigv, bnw + layer * 64, bnb + layer * 64, rightP,
        swOmW1[layer], omb1 + layer * 64, swOmW2[layer], omb2 + layer * 64, outP);
  };
  auto run_gn = [&](float* xP, int layer) {
    colsum64_kernel<<<RB, 256, 0, stream>>>(xP, NNODE, part);
    finalize_norm_kernel<<<1, 64, 0, stream>>>(part, RB, NNODE, 1, gnA + layer * 64, meanv, rsigv);
    apply_gn_kernel<<<gridElems, 256, 0, stream>>>(xP, NNODE, meanv, rsigv,
                                                   gnW + layer * 64, gnB + layer * 64,
                                                   gnA + layer * 64);
  };

  float* cf = cfA; float* cft = cfB;
  float* vf = vfA; float* vft = vfB;

  // ---- depth 0 (layers 0: v->c, 1: c->v) ----
  run_conv(vf, cf, eV, eC, cntC, 0, cft); { float* t = cf; cf = cft; cft = t; }
  run_conv(cf, vf, eC, eV, cntV, 1, vft); { float* t = vf; vf = vft; vft = t; }
  run_gn(cf, 0);
  run_gn(vf, 1);
  float* vf1 = vf;  // vars_outputs[1] (== vfB, preserved hereafter)

  // ---- depth 1 (layers 2: v->c, 3: c->v) ----
  run_conv(vf, cf, eV, eC, cntC, 2, cft); { float* t = cf; cf = cft; cft = t; }
  run_conv(cf, vf, eC, eV, cntV, 3, vft); { float* t = vf; vf = vft; vft = t; }
  run_gn(cf, 2);
  run_gn(vf, 3);  // vf == vars_outputs[2]

  // ---- output MLP ----
  final_mlp_kernel<<<NNODE / 16, 256, 0, stream>>>(vf0, vf1, vf, swOW1, ob1, swOW2, ob2,
                                                   oW3, ob3, (float*)d_out);
}
